// Attention_670014898233
// MI455X (gfx1250) — matrix operand order repaired and checked by reading, not compiled, not run on hardware
//
#include <hip/hip_runtime.h>
#include <hip/hip_bf16.h>

typedef __attribute__((ext_vector_type(16))) __bf16 v16bf;
typedef __attribute__((ext_vector_type(8)))  __bf16 v8bf;
typedef __attribute__((ext_vector_type(4)))  __bf16 v4bf;
typedef __attribute__((ext_vector_type(8)))  float  v8f;
typedef __attribute__((ext_vector_type(4)))  unsigned v4u;
typedef __attribute__((ext_vector_type(8)))  int    v8i;
typedef __attribute__((ext_vector_type(4)))  int    v4i;

#define NXDIM 1024
#define NHEAD 16
#define HDIM  64
#define SEQ   2048
#define NROWS 4096   // B*S = 2*2048

// ---------------------------------------------------------------------------
// helpers
// ---------------------------------------------------------------------------
static __device__ __forceinline__ v16bf cat8(v8bf lo, v8bf hi) {
  return __builtin_shufflevector(lo, hi, 0,1,2,3,4,5,6,7,8,9,10,11,12,13,14,15);
}

// A-fragment (16x32 bf16): lane L<16 holds row L, K=0..7 (v0-3) & K=16..23 (v4-7);
// lane L+16 holds row L, K=8..15 & 24..31.
static __device__ __forceinline__ v16bf load_afrag(const __bf16* base, int stride, int lane) {
  const int row = lane & 15;
  const int kb  = (lane >> 4) * 8;
  const __bf16* p = base + row * stride + kb;
  v8bf lo = *(const v8bf*)p;
  v8bf hi = *(const v8bf*)(p + 16);
  return cat8(lo, hi);
}

// B-fragment (32x16 bf16), source stored N-major (row n = column of B, 32 contiguous K):
// lane L<16 holds col L, K=0..7 & 16..23; lane L+16 holds col L, K=8..15 & 24..31.
static __device__ __forceinline__ v16bf load_bfrag(const __bf16* base, int stride, int lane) {
  const int n  = lane & 15;
  const int kb = (lane >> 4) * 8;
  const __bf16* p = base + n * stride + kb;
  v8bf lo = *(const v8bf*)p;
  v8bf hi = *(const v8bf*)(p + 16);
  return cat8(lo, hi);
}

static __device__ __forceinline__ v8f wmma_bf16(v16bf a, v16bf b, v8f c) {
  return __builtin_amdgcn_wmma_f32_16x16x32_bf16(false, a, false, b, (short)0, c,
                                                 false, false);
}

static __device__ __forceinline__ float redmax16(float v) {
#pragma unroll
  for (int m = 1; m < 16; m <<= 1) v = fmaxf(v, __shfl_xor(v, m, 32));
  return v;
}
static __device__ __forceinline__ float redsum16(float v) {
#pragma unroll
  for (int m = 1; m < 16; m <<= 1) v += __shfl_xor(v, m, 32);
  return v;
}

// raw 32-bit LDS offset of a __shared__ object (addrspace(3) ptrtoint)
static __device__ __forceinline__ unsigned lds_addr32(void* p) {
  return (unsigned)(unsigned long long)(__attribute__((address_space(3))) char*)p;
}

// ---------------------------------------------------------------------------
// Tensor Data Mover: async load of a 128x32-bf16 tile (row stride 1024 bf16)
// into LDS with hardware padding 64B->80B per row ([128][40] layout).
// D# per CDNA5 ISA ch.8: group0 = {count=1 | lds_addr | global_addr | type=2},
// group1 = {data_size=2B, pad_enable, pad_interval=16dw, pad_amount=4dw,
//           tensor_dim0 = 2^30+2^14, tensor_dim1 = 2^30 (never OOB), tile=32x128, stride0=1024}.
// ---------------------------------------------------------------------------
static __device__ __forceinline__ void tdm_load_tile_128x32(unsigned ldsoff,
                                                            const void* gptr) {
  const unsigned long long ga = (unsigned long long)gptr;
  v4u g0;
  g0[0] = 1u;                                   // count=1, user descriptor
  g0[1] = ldsoff;                               // lds_addr (bytes)
  g0[2] = (unsigned)ga;                         // global_addr[31:0]
  g0[3] = (unsigned)((ga >> 32) & 0x01FFFFFFu) | 0x80000000u;  // addr[56:32] | type=2
  v8i g1;
  g1[0] = (int)((1u << 16) |                    // data_size = 2 bytes
                (1u << 20) |                    // pad_enable
                (3u << 22) |                    // pad_interval: 16 dwords (64B)
                (3u << 25));                    // pad_amount: 4 dwords (16B)
  g1[1] = (int)((1u << 30) & 0xFFFF0000u);      // atomic_addr=0 | tensor_dim0[15:0]=0x4000
  g1[2] = (int)(0x4000u |                       // tensor_dim0[31:16] (2^30)
                0x0000u);                       // tensor_dim1[15:0]=0
  g1[3] = (int)(0x4000u |                       // tensor_dim1[31:16] (2^30)
                (32u << 16));                   // tile_dim0 = 32 elements
  g1[4] = 128;                                  // tile_dim1 = 128, tile_dim2 = 0
  g1[5] = 1024;                                 // tensor_dim0_stride[31:0] = 1024
  g1[6] = 0;
  g1[7] = 0;
  v4i z4 = {0, 0, 0, 0};
#if __clang_major__ >= 23
  v8i z8 = {0, 0, 0, 0, 0, 0, 0, 0};
  __builtin_amdgcn_tensor_load_to_lds(g0, g1, z4, z4, z8, 0);
#else
  __builtin_amdgcn_tensor_load_to_lds(g0, g1, z4, z4, 0);
#endif
}

// ---------------------------------------------------------------------------
// fp32 -> bf16 convert / transpose-convert
// ---------------------------------------------------------------------------
__global__ void cvt_f32_bf16(const float* __restrict__ in, __bf16* __restrict__ out, int n) {
  int i = (blockIdx.x * 256 + threadIdx.x) * 4;
  if (i >= n) return;
  const float4 f = *(const float4*)(in + i);
  v4bf o;
  o[0] = (__bf16)f.x; o[1] = (__bf16)f.y; o[2] = (__bf16)f.z; o[3] = (__bf16)f.w;
  *(v4bf*)(out + i) = o;
}

// in: [kdim][nrows]  ->  out: [nrows][kdim]  (bf16, N-major for WMMA B)
__global__ void transpose_f32_bf16(const float* __restrict__ in, __bf16* __restrict__ out,
                                   int nrows, int kdim) {
  int t = blockIdx.x * 256 + threadIdx.x;
  if (t >= nrows * kdim) return;
  int n = t / kdim;
  int k = t - n * kdim;
  out[t] = (__bf16)in[(size_t)k * nrows + n];
}

// ---------------------------------------------------------------------------
// QKV GEMM: [4096,1024](bf16) x [1024,3072] (WT N-major bf16) + bias
//   -> Q [B,H,S,D]*0.125, K [B,H,S,D], V^T [B,H,D,S]   (all bf16)
// 128x128 block, 8 waves, wave tile 32x64, K-step 32.
// TDM double-buffered global->LDS staging (wave 0 drives the DMA).
// ---------------------------------------------------------------------------
__global__ __launch_bounds__(256) void qkv_gemm(const __bf16* __restrict__ X,
                                                const __bf16* __restrict__ WT,
                                                const float*  __restrict__ bias,
                                                __bf16* __restrict__ Q,
                                                __bf16* __restrict__ K,
                                                __bf16* __restrict__ VT) {
  __shared__ alignas(16) __bf16 As[2][128][40];
  __shared__ alignas(16) __bf16 Bs[2][128][40];
  const int tid  = threadIdx.x;
  const int lane = tid & 31;
  const int w    = tid >> 5;
  const int wm   = w >> 1;          // 0..3
  const int wn   = w & 1;           // 0..1
  const int m0   = blockIdx.y * 128;
  const int n0   = blockIdx.x * 128;
  const __bf16* Ag = X  + (size_t)m0 * NXDIM;
  const __bf16* Bg = WT + (size_t)n0 * NXDIM;

  if (w == 0) {
    tdm_load_tile_128x32(lds_addr32(&As[0][0][0]), Ag);
    tdm_load_tile_128x32(lds_addr32(&Bs[0][0][0]), Bg);
  }

  v8f acc[2][4];
#pragma unroll
  for (int mi = 0; mi < 2; ++mi)
#pragma unroll
    for (int ni = 0; ni < 4; ++ni) acc[mi][ni] = (v8f)(0.0f);

  for (int kb = 0; kb < 32; ++kb) {
    const int cur = kb & 1;
    if (kb + 1 < 32) {
      if (w == 0) {
        tdm_load_tile_128x32(lds_addr32(&As[cur ^ 1][0][0]), Ag + (kb + 1) * 32);
        tdm_load_tile_128x32(lds_addr32(&Bs[cur ^ 1][0][0]), Bg + (kb + 1) * 32);
        __builtin_amdgcn_s_wait_tensorcnt(2);   // current tile's two DMAs done
      }
    } else if (w == 0) {
      __builtin_amdgcn_s_wait_tensorcnt(0);
    }
    __syncthreads();
    asm volatile("" ::: "memory");
    v16bf af[2], bf[4];
#pragma unroll
    for (int mi = 0; mi < 2; ++mi) af[mi] = load_afrag(&As[cur][wm * 32 + mi * 16][0], 40, lane);
#pragma unroll
    for (int ni = 0; ni < 4; ++ni) bf[ni] = load_bfrag(&Bs[cur][wn * 64 + ni * 16][0], 40, lane);
#pragma unroll
    for (int mi = 0; mi < 2; ++mi)
#pragma unroll
      for (int ni = 0; ni < 4; ++ni)
        acc[mi][ni] = wmma_bf16(af[mi], bf[ni], acc[mi][ni]);
    __syncthreads();
  }

  const int colL   = lane & 15;
  const int rowoff = (lane < 16) ? 0 : 8;
#pragma unroll
  for (int mi = 0; mi < 2; ++mi)
#pragma unroll
    for (int ni = 0; ni < 4; ++ni) {
      const int n    = n0 + wn * 64 + ni * 16 + colL;
      const int sect = n >> 10;            // 0=Q 1=K 2=V (uniform per wave)
      const int c    = n & 1023;
      const int h    = c >> 6;
      const int d    = c & 63;
      const float bv = bias[n];
      if (sect == 0) {
#pragma unroll
        for (int r = 0; r < 8; ++r) {
          const int m = m0 + wm * 32 + mi * 16 + rowoff + r;
          const size_t head = (size_t)((m >> 11) * NHEAD + h);
          Q[(head * SEQ + (m & 2047)) * HDIM + d] = (__bf16)((acc[mi][ni][r] + bv) * 0.125f);
        }
      } else if (sect == 1) {
#pragma unroll
        for (int r = 0; r < 8; ++r) {
          const int m = m0 + wm * 32 + mi * 16 + rowoff + r;
          const size_t head = (size_t)((m >> 11) * NHEAD + h);
          K[(head * SEQ + (m & 2047)) * HDIM + d] = (__bf16)(acc[mi][ni][r] + bv);
        }
      } else {
#pragma unroll
        for (int r = 0; r < 8; ++r) {
          const int m = m0 + wm * 32 + mi * 16 + rowoff + r;
          const size_t head = (size_t)((m >> 11) * NHEAD + h);
          VT[(head * HDIM + d) * SEQ + (m & 2047)] = (__bf16)(acc[mi][ni][r] + bv);
        }
      }
    }
}

// ---------------------------------------------------------------------------
// Causal flash attention: one wave per 16-row Q tile, 32-key blocks.
// Q pre-scaled by 1/sqrt(64). Output A (merged heads) bf16 [4096,1024].
// ---------------------------------------------------------------------------
__global__ __launch_bounds__(128) void attn_kernel(const __bf16* __restrict__ Q,
                                                   const __bf16* __restrict__ K,
                                                   const __bf16* __restrict__ VT,
                                                   __bf16* __restrict__ A) {
  __shared__ alignas(16) __bf16 P[4][16][40];   // per-wave P tile 16x32 (padded)
  const int w    = threadIdx.x >> 5;
  const int lane = threadIdx.x & 31;
  const int qt   = blockIdx.x * 4 + w;          // 0..127
  const int bh   = blockIdx.y;                  // 0..31
  const int m0   = qt * 16;

  const __bf16* Qh = Q  + (size_t)bh * SEQ * HDIM;
  const __bf16* Kh = K  + (size_t)bh * SEQ * HDIM;
  const __bf16* Vh = VT + (size_t)bh * HDIM * SEQ;
  __bf16 (*Pw)[40] = P[w];

  v16bf qf[2];
#pragma unroll
  for (int t = 0; t < 2; ++t)
    qf[t] = load_afrag(Qh + (size_t)m0 * HDIM + t * 32, HDIM, lane);

  v8f o[4];
#pragma unroll
  for (int t = 0; t < 4; ++t) o[t] = (v8f)(0.0f);
  float Mrun[8], Lrun[8];
#pragma unroll
  for (int r = 0; r < 8; ++r) { Mrun[r] = -3.0e38f; Lrun[r] = 0.0f; }

  const int rowoff = (lane < 16) ? 0 : 8;
  const int col    = lane & 15;
  const int nblk   = (m0 + 16 + 31) >> 5;       // causal key-block count

  for (int jb = 0; jb < nblk; ++jb) {
    const int j0 = jb * 32;
    v8f s0 = (v8f)(0.0f), s1 = (v8f)(0.0f);
#pragma unroll
    for (int t = 0; t < 2; ++t) {               // contraction over D in 2x32
      v16bf kf0 = load_bfrag(Kh + (size_t)(j0)      * HDIM + t * 32, HDIM, lane);
      v16bf kf1 = load_bfrag(Kh + (size_t)(j0 + 16) * HDIM + t * 32, HDIM, lane);
      s0 = wmma_bf16(qf[t], kf0, s0);
      s1 = wmma_bf16(qf[t], kf1, s1);
    }
    // online softmax over the 32-key block
#pragma unroll
    for (int r = 0; r < 8; ++r) {
      const int m = m0 + rowoff + r;
      float e0 = (j0 + col      <= m) ? s0[r] : -10000.0f;
      float e1 = (j0 + 16 + col <= m) ? s1[r] : -10000.0f;
      float tmax = redmax16(fmaxf(e0, e1));
      float newM = fmaxf(Mrun[r], tmax);
      float alpha = __expf(Mrun[r] - newM);
      float p0 = __expf(e0 - newM);
      float p1 = __expf(e1 - newM);
      float rs = redsum16(p0 + p1);
      Lrun[r] = Lrun[r] * alpha + rs;
      Mrun[r] = newM;
#pragma unroll
      for (int t = 0; t < 4; ++t) o[t][r] *= alpha;
      Pw[rowoff + r][col]      = (__bf16)p0;
      Pw[rowoff + r][16 + col] = (__bf16)p1;
    }
    asm volatile("s_wait_dscnt 0x0" ::: "memory");   // stores -> cross-lane reads
    v16bf pf = load_afrag(&Pw[0][0], 40, lane);      // P as 16x32 A-fragment
#pragma unroll
    for (int t = 0; t < 4; ++t) {
      v16bf vf = load_bfrag(Vh + (size_t)(t * 16) * SEQ + j0, SEQ, lane);
      o[t] = wmma_bf16(pf, vf, o[t]);
    }
  }

  const int b = bh >> 4, h = bh & 15;
#pragma unroll
  for (int r = 0; r < 8; ++r) {
    const float inv = 1.0f / Lrun[r];
    const int   s   = m0 + rowoff + r;
    __bf16* rowp = A + ((size_t)(b * SEQ + s)) * NXDIM + h * HDIM;
#pragma unroll
    for (int t = 0; t < 4; ++t) rowp[t * 16 + col] = (__bf16)(o[t][r] * inv);
  }
}

// ---------------------------------------------------------------------------
// Output projection: [4096,1024](bf16) x [1024,1024](WT N-major bf16) + bias -> fp32
// ---------------------------------------------------------------------------
__global__ __launch_bounds__(256) void proj_gemm(const __bf16* __restrict__ Ain,
                                                 const __bf16* __restrict__ WT,
                                                 const float*  __restrict__ bias,
                                                 float* __restrict__ out) {
  __shared__ alignas(16) __bf16 As[2][128][40];
  __shared__ alignas(16) __bf16 Bs[2][128][40];
  const int tid  = threadIdx.x;
  const int lane = tid & 31;
  const int w    = tid >> 5;
  const int wm   = w >> 1;
  const int wn   = w & 1;
  const int m0   = blockIdx.y * 128;
  const int n0   = blockIdx.x * 128;
  const __bf16* Ag = Ain + (size_t)m0 * NXDIM;
  const __bf16* Bg = WT  + (size_t)n0 * NXDIM;

  if (w == 0) {
    tdm_load_tile_128x32(lds_addr32(&As[0][0][0]), Ag);
    tdm_load_tile_128x32(lds_addr32(&Bs[0][0][0]), Bg);
  }

  v8f acc[2][4];
#pragma unroll
  for (int mi = 0; mi < 2; ++mi)
#pragma unroll
    for (int ni = 0; ni < 4; ++ni) acc[mi][ni] = (v8f)(0.0f);

  for (int kb = 0; kb < 32; ++kb) {
    const int cur = kb & 1;
    if (kb + 1 < 32) {
      if (w == 0) {
        tdm_load_tile_128x32(lds_addr32(&As[cur ^ 1][0][0]), Ag + (kb + 1) * 32);
        tdm_load_tile_128x32(lds_addr32(&Bs[cur ^ 1][0][0]), Bg + (kb + 1) * 32);
        __builtin_amdgcn_s_wait_tensorcnt(2);
      }
    } else if (w == 0) {
      __builtin_amdgcn_s_wait_tensorcnt(0);
    }
    __syncthreads();
    asm volatile("" ::: "memory");
    v16bf af[2], bf[4];
#pragma unroll
    for (int mi = 0; mi < 2; ++mi) af[mi] = load_afrag(&As[cur][wm * 32 + mi * 16][0], 40, lane);
#pragma unroll
    for (int ni = 0; ni < 4; ++ni) bf[ni] = load_bfrag(&Bs[cur][wn * 64 + ni * 16][0], 40, lane);
#pragma unroll
    for (int mi = 0; mi < 2; ++mi)
#pragma unroll
      for (int ni = 0; ni < 4; ++ni)
        acc[mi][ni] = wmma_bf16(af[mi], bf[ni], acc[mi][ni]);
    __syncthreads();
  }

  const int colL   = lane & 15;
  const int rowoff = (lane < 16) ? 0 : 8;
#pragma unroll
  for (int mi = 0; mi < 2; ++mi)
#pragma unroll
    for (int ni = 0; ni < 4; ++ni) {
      const int n = n0 + wn * 64 + ni * 16 + colL;
      const float bv = bias[n];
#pragma unroll
      for (int r = 0; r < 8; ++r) {
        const int m = m0 + wm * 32 + mi * 16 + rowoff + r;
        out[(size_t)m * NXDIM + n] = acc[mi][ni][r] + bv;
      }
    }
}

// ---------------------------------------------------------------------------
// launch
// ---------------------------------------------------------------------------
extern "C" void kernel_launch(void* const* d_in, const int* in_sizes, int n_in,
                              void* d_out, int out_size, void* d_ws, size_t ws_size,
                              hipStream_t stream) {
  (void)in_sizes; (void)n_in; (void)out_size; (void)ws_size;
  const float* hs = (const float*)d_in[0];   // [2,2048,1024]
  const float* aw = (const float*)d_in[1];   // [1024,3072]
  const float* ab = (const float*)d_in[2];   // [3072]
  const float* pw = (const float*)d_in[3];   // [1024,1024]
  const float* pb = (const float*)d_in[4];   // [1024]
  float* out = (float*)d_out;

  char* ws = (char*)d_ws;
  __bf16* Xbf = (__bf16*)(ws + 0);                    //  8 MB  [4096,1024]
  __bf16* WAt = (__bf16*)(ws + 8388608);              //  6 MB  [3072,1024]
  __bf16* WPt = (__bf16*)(ws + 14680064);             //  2 MB  [1024,1024]
  __bf16* Qb  = (__bf16*)(ws + 16777216);             //  8 MB  [B,H,S,D]
  __bf16* Kb  = (__bf16*)(ws + 25165824);             //  8 MB  [B,H,S,D]
  __bf16* VTb = (__bf16*)(ws + 33554432);             //  8 MB  [B,H,D,S]
  __bf16* Ab  = (__bf16*)(ws + 41943040);             //  8 MB  [4096,1024]

  cvt_f32_bf16<<<4096, 256, 0, stream>>>(hs, Xbf, NROWS * NXDIM);
  transpose_f32_bf16<<<12288, 256, 0, stream>>>(aw, WAt, 3 * NXDIM, NXDIM);
  transpose_f32_bf16<<<4096, 256, 0, stream>>>(pw, WPt, NXDIM, NXDIM);

  qkv_gemm<<<dim3(24, 32), 256, 0, stream>>>(Xbf, WAt, ab, Qb, Kb, VTb);
  attn_kernel<<<dim3(32, 32), 128, 0, stream>>>(Qb, Kb, VTb, Ab);
  proj_gemm<<<dim3(8, 32), 256, 0, stream>>>(Ab, WPt, pb, out);
}
